// TPCausalSelfAttention_35828617183754
// MI455X (gfx1250) — matrix operand order repaired and checked by reading, not compiled, not run on hardware
//
#include <hip/hip_runtime.h>
#include <hip/hip_bf16.h>

#define D_MODEL 1024
#define SEQ     2048
#define NB      2
#define NH      16
#define HD      64
#define MROWS   (NB * SEQ)   // 4096

typedef __attribute__((ext_vector_type(16))) __bf16    v16bf;
typedef __attribute__((ext_vector_type(8)))  float     v8f;
typedef __attribute__((ext_vector_type(4)))  unsigned  v4u;
typedef __attribute__((ext_vector_type(8)))  int       v8i;
typedef __attribute__((ext_vector_type(4)))  int       v4i;

static __device__ inline v8f wmma_bf16(v16bf a, v16bf b, v8f c) {
  return __builtin_amdgcn_wmma_f32_16x16x32_bf16(
      /*neg_a=*/false, a, /*neg_b=*/false, b,
      /*c_mod=*/(short)0, c, /*reuse_a=*/false, /*reuse_b=*/false);
}

// pack two floats -> two bf16 in one dword (for 4B-aligned LDS stores)
static __device__ inline unsigned pack2(float a, float b) {
  union { unsigned u; __bf16 h[2]; } z;
  z.h[0] = (__bf16)a; z.h[1] = (__bf16)b;
  return z.u;
}

// ---------------------------------------------------------------------------
// Tensor Data Mover: 2D tile load Global -> LDS (bf16 elements, data_size=2B)
// D# built per CDNA5 ISA 8.3/8.4: group0 {count=1, lds_addr, global_addr,
// type=2}; group1 {data_size=1(2B), tensor_dim0/1, tile_dim0/1, dim0_stride}.
// Tracked by TENSORcnt; synchronized with s_wait_tensorcnt.
// ---------------------------------------------------------------------------
#if defined(__clang_major__) && (__clang_major__ >= 23)
static __device__ inline void tdm_call(v4u g0, v8i g1, v4i g2, v4i g3) {
  v8i z = {};
  __builtin_amdgcn_tensor_load_to_lds(g0, g1, g2, g3, z, 0);
}
#else
static __device__ inline void tdm_call(v4u g0, v8i g1, v4i g2, v4i g3) {
  __builtin_amdgcn_tensor_load_to_lds(g0, g1, g2, g3, 0);
}
#endif

static __device__ inline void tdm_load_tile_2d(
    unsigned lds_addr, const void* gaddr,
    unsigned tensor_d0, unsigned tensor_d1,
    unsigned tile_d0, unsigned tile_d1,
    unsigned stride_d0 /* elements */) {
  unsigned long long ga = (unsigned long long)(uintptr_t)gaddr;
  v4u g0;
  g0[0] = 1u;                                           // count=1 (valid user D#)
  g0[1] = lds_addr;                                     // LDS byte address
  g0[2] = (unsigned)(ga & 0xFFFFFFFFu);                 // global_addr[31:0]
  g0[3] = (unsigned)((ga >> 32) & 0x01FFFFFFu)          // global_addr[56:32]
          | (2u << 30);                                 // type=2 ("image")
  v8i g1;
  g1[0] = (int)(1u << 16);                              // data_size=1 -> 2 bytes
  g1[1] = (int)((tensor_d0 & 0xFFFFu) << 16);           // tensor_dim0[15:0]
  g1[2] = (int)((tensor_d0 >> 16) |
                ((tensor_d1 & 0xFFFFu) << 16));         // dim0 hi | dim1 lo
  g1[3] = (int)((tensor_d1 >> 16) | (tile_d0 << 16));   // dim1 hi | tile_dim0
  g1[4] = (int)(tile_d1 & 0xFFFFu);                     // tile_dim1 (tile_dim2=0)
  g1[5] = (int)stride_d0;                               // tensor_dim0_stride lo32
  g1[6] = 0;                                            // stride hi | dim1_stride lo
  g1[7] = 0;
  v4i gz = {};
  tdm_call(g0, g1, gz, gz);
}

// A-fragment (16x32 bf16, MxK), per ISA 7.12.2:
// lane: M = lane&15 ; half = lane>>4 ; VGPR vg holds K pair
//   k(vg) = (vg<4 ? 0 : 16) + half*8 + 2*(vg&3)
static __device__ inline v16bf load_a_frag(const __bf16* tile, int ldk, int lane) {
  const int half = lane >> 4;
  const int m    = lane & 15;
  v16bf a;
#pragma unroll
  for (int vg = 0; vg < 8; ++vg) {
    const int k = ((vg < 4) ? 0 : 16) + half * 8 + 2 * (vg & 3);
    a[2 * vg]     = tile[m * ldk + k];
    a[2 * vg + 1] = tile[m * ldk + k + 1];
  }
  return a;
}

// B-fragment (32x16 bf16, KxN), N-major source: elem(k,n) = tile[n*ldk + k]
static __device__ inline v16bf load_b_frag(const __bf16* tile, int ldk, int lane) {
  const int n     = lane & 15;
  const int khalf = (lane >> 4) * 8;
  v16bf b;
#pragma unroll
  for (int vg = 0; vg < 8; ++vg) {
    const int k = ((vg < 4) ? 0 : 16) + khalf + 2 * (vg & 3);
    b[2 * vg]     = tile[n * ldk + k];
    b[2 * vg + 1] = tile[n * ldk + k + 1];
  }
  return b;
}

// B-fragment from K-major source (e.g. V tile [key][feat]): elem(k,n)=tile[k*ldn + noff + n]
static __device__ inline v16bf load_b_frag_T(const __bf16* tile, int ldn, int noff, int lane) {
  const int n     = lane & 15;
  const int khalf = (lane >> 4) * 8;
  v16bf b;
#pragma unroll
  for (int vg = 0; vg < 8; ++vg) {
    const int k = ((vg < 4) ? 0 : 16) + khalf + 2 * (vg & 3);
    b[2 * vg]     = tile[k * ldn + noff + n];
    b[2 * vg + 1] = tile[(k + 1) * ldn + noff + n];
  }
  return b;
}

// ---------------------------------------------------------------------------
// Kernel 1: qkv = x @ w_attn^T + b_attn -> Q/K/V bf16 in [B,H,S,64] layout.
// Q pre-scaled by 1/sqrt(hd)=0.125.  128 threads / 4 waves, 64x128 tile,
// K stepped by 32: per wave 1 A-frag + 8 B-frags -> 8 WMMAs.
// ---------------------------------------------------------------------------
__global__ __launch_bounds__(128) void qkv_gemm_kernel(
    const float* __restrict__ x, const float* __restrict__ w_attn,
    const float* __restrict__ b_attn, __bf16* __restrict__ Qws,
    __bf16* __restrict__ Kws, __bf16* __restrict__ Vws) {
  __shared__ __align__(16) __bf16 As[64 * 32];
  __shared__ __align__(16) __bf16 Bs[128 * 32];

  const int tid  = threadIdx.x;
  const int lane = tid & 31;
  const int wave = tid >> 5;
  const int gm0  = blockIdx.x * 64;
  const int gn0  = blockIdx.y * 128;

  const int arow = tid >> 1;         // 0..63
  const int acol = (tid & 1) * 16;   // 0 or 16

  v8f acc[8] = {};

  for (int k0 = 0; k0 < D_MODEL; k0 += 32) {
    const float* aSrc = x      + (size_t)(gm0 + arow) * D_MODEL + k0 + acol;
    const float* bSrc = w_attn + (size_t)(gn0 + tid) * D_MODEL + k0;
    if (k0 + 32 < D_MODEL) {
      __builtin_prefetch(aSrc + 32, 0, 1);
      __builtin_prefetch(bSrc + 32, 0, 1);
    }
    // A tile: 64x32 fp32 -> bf16 (16 floats/thread)
#pragma unroll
    for (int i = 0; i < 4; ++i) {
      float4 av = ((const float4*)aSrc)[i];
      const int di = arow * 32 + acol + 4 * i;
      *(unsigned*)&As[di]     = pack2(av.x, av.y);
      *(unsigned*)&As[di + 2] = pack2(av.z, av.w);
    }
    // B tile: 128x32 fp32 -> bf16 (one full row/thread)
#pragma unroll
    for (int i = 0; i < 8; ++i) {
      float4 bv = ((const float4*)bSrc)[i];
      const int di = tid * 32 + 4 * i;
      *(unsigned*)&Bs[di]     = pack2(bv.x, bv.y);
      *(unsigned*)&Bs[di + 2] = pack2(bv.z, bv.w);
    }
    __syncthreads();

    const v16bf afrag = load_a_frag(&As[wave * 16 * 32], 32, lane);
#pragma unroll
    for (int j = 0; j < 8; ++j) {
      const v16bf bfrag = load_b_frag(&Bs[j * 16 * 32], 32, lane);
      acc[j] = wmma_bf16(afrag, bfrag, acc[j]);
    }
    __syncthreads();
  }

  // Scatter + bias into Q/K/V workspace
  const int half = lane >> 4;
  const int coln = lane & 15;
#pragma unroll
  for (int j = 0; j < 8; ++j) {
#pragma unroll
    for (int r = 0; r < 8; ++r) {
      const int gm = gm0 + wave * 16 + half * 8 + r;   // global row (b*S + s)
      const int gn = gn0 + j * 16 + coln;              // global col in [0,3072)
      float v = acc[j][r] + b_attn[gn];
      const int t    = gn >> 10;        // 0=Q 1=K 2=V
      const int dloc = gn & 1023;
      const int h    = dloc >> 6;
      const int f    = dloc & 63;
      const int b    = gm >> 11;
      const int s    = gm & 2047;
      if (t == 0) v *= 0.125f;          // fold 1/sqrt(hd) into Q
      __bf16* dst = (t == 0) ? Qws : (t == 1) ? Kws : Vws;
      dst[((size_t)(b * NH + h) * SEQ + s) * HD + f] = (__bf16)v;
    }
  }
}

// ---------------------------------------------------------------------------
// Kernel 2: causal flash attention per (b,h). 128 threads = 4 waves.
// Block handles 64 q rows; each wave a 16-row q tile. KV tiles of 64 keys,
// staged Global->LDS by the Tensor Data Mover with DOUBLE BUFFERING:
// tile t+1's DMA is issued before computing tile t, then s_wait_tensorcnt(2)
// retires tile t's two in-order loads while t+1's remain in flight.
// ---------------------------------------------------------------------------
__global__ __launch_bounds__(128) void flash_attn_kernel(
    const __bf16* __restrict__ Qws, const __bf16* __restrict__ Kws,
    const __bf16* __restrict__ Vws, __bf16* __restrict__ Yattn) {
  __shared__ __align__(16) __bf16 Kt[2][64 * 64];
  __shared__ __align__(16) __bf16 Vt[2][64 * 64];
  __shared__ __align__(16) __bf16 Pscr[4][16 * 32];

  const int tid  = threadIdx.x;
  const int lane = tid & 31;
  const int wave = tid >> 5;
  const int half = lane >> 4;
  const int coln = lane & 15;

  const int qTile   = blockIdx.x;       // 0..31
  const int h       = blockIdx.y;
  const int b       = blockIdx.z;
  const int bh      = b * NH + h;
  const int blockQ0 = qTile * 64;
  const int qBase   = blockQ0 + wave * 16;

  const __bf16* Qbh = Qws + (size_t)bh * SEQ * HD;
  const __bf16* Kbh = Kws + (size_t)bh * SEQ * HD;
  const __bf16* Vbh = Vws + (size_t)bh * SEQ * HD;

  unsigned ktLds[2], vtLds[2];
  ktLds[0] = (unsigned)(uintptr_t)(void*)&Kt[0][0];
  ktLds[1] = (unsigned)(uintptr_t)(void*)&Kt[1][0];
  vtLds[0] = (unsigned)(uintptr_t)(void*)&Vt[0][0];
  vtLds[1] = (unsigned)(uintptr_t)(void*)&Vt[1][0];

  // Q fragments: 16x64 split as two 16x32 K-chunks (A-fragment layout)
  v16bf qf[2];
#pragma unroll
  for (int c = 0; c < 2; ++c) {
#pragma unroll
    for (int vg = 0; vg < 8; ++vg) {
      const int k = 32 * c + ((vg < 4) ? 0 : 16) + half * 8 + 2 * (vg & 3);
      qf[c][2 * vg]     = Qbh[(size_t)(qBase + coln) * HD + k];
      qf[c][2 * vg + 1] = Qbh[(size_t)(qBase + coln) * HD + k + 1];
    }
  }

  float mi[8], li[8];
  v8f O[4] = {};
#pragma unroll
  for (int r = 0; r < 8; ++r) { mi[r] = -3.0e38f; li[r] = 0.0f; }

  const int tMax = blockQ0 / 64;      // inclusive causal tile bound

  // Prologue: kick off tile 0 DMA into buffer 0
  if (wave == 0) {
    tdm_load_tile_2d(ktLds[0], Kbh, HD, SEQ, HD, 64, HD);
    tdm_load_tile_2d(vtLds[0], Vbh, HD, SEQ, HD, 64, HD);
  }

  for (int t = 0; t <= tMax; ++t) {
    const int cur = t & 1;
    if (wave == 0) {
      if (t < tMax) {
        // issue-ahead: tile t+1 into the other buffer, then retire tile t
        const int nxt = cur ^ 1;
        tdm_load_tile_2d(ktLds[nxt], Kbh + (size_t)((t + 1) * 64) * HD,
                         HD, SEQ, HD, 64, HD);
        tdm_load_tile_2d(vtLds[nxt], Vbh + (size_t)((t + 1) * 64) * HD,
                         HD, SEQ, HD, 64, HD);
        __builtin_amdgcn_s_wait_tensorcnt(2);   // tile t done; t+1 in flight
      } else {
        __builtin_amdgcn_s_wait_tensorcnt(0);   // drain
      }
    }
    __syncthreads();

    const __bf16* Kc = &Kt[cur][0];
    const __bf16* Vc = &Vt[cur][0];

#pragma unroll
    for (int sub = 0; sub < 2; ++sub) {
      const int k0 = t * 64 + sub * 32;      // global key start of sub-step
      if (k0 <= qBase + 15) {                // wave-uniform causal guard
        // ---- scores: two 16x16 tiles, feature dim 64 = 2 x K32 ----
        v8f s0 = {}, s1 = {};
        s0 = wmma_bf16(qf[0], load_b_frag(&Kc[(sub * 32) * 64],           64, lane), s0);
        s0 = wmma_bf16(qf[1], load_b_frag(&Kc[(sub * 32) * 64 + 32],      64, lane), s0);
        s1 = wmma_bf16(qf[0], load_b_frag(&Kc[(sub * 32 + 16) * 64],      64, lane), s1);
        s1 = wmma_bf16(qf[1], load_b_frag(&Kc[(sub * 32 + 16) * 64 + 32], 64, lane), s1);

        const bool edge = (k0 + 31 > qBase);   // tile touches the diagonal
#pragma unroll
        for (int r = 0; r < 8; ++r) {
          const int qg = qBase + half * 8 + r;
          if (edge) {
            if (k0 + coln > qg)      s0[r] = -3.0e38f;
            if (k0 + 16 + coln > qg) s1[r] = -3.0e38f;
          }
          // online softmax: row reductions across 16-lane half-groups
          float rmax = fmaxf(s0[r], s1[r]);
#pragma unroll
          for (int off = 8; off; off >>= 1)
            rmax = fmaxf(rmax, __shfl_xor(rmax, off, 16));
          const float mnew  = fmaxf(mi[r], rmax);
          const float alpha = __expf(mi[r] - mnew);
          const float p0 = __expf(s0[r] - mnew);
          const float p1 = __expf(s1[r] - mnew);
          float rsum = p0 + p1;
#pragma unroll
          for (int off = 8; off; off >>= 1)
            rsum += __shfl_xor(rsum, off, 16);
          li[r] = li[r] * alpha + rsum;
          mi[r] = mnew;
#pragma unroll
          for (int j = 0; j < 4; ++j) O[j][r] *= alpha;
          // stash P (16x32) in A-fragment-readable LDS scratch
          const int prow = half * 8 + r;
          Pscr[wave][prow * 32 + coln]      = (__bf16)p0;
          Pscr[wave][prow * 32 + 16 + coln] = (__bf16)p1;
        }

        // ---- O += P (16x32) x V (32x64) ----
        const v16bf pf = load_a_frag(&Pscr[wave][0], 32, lane);
#pragma unroll
        for (int j = 0; j < 4; ++j) {
          const v16bf bv = load_b_frag_T(&Vc[(sub * 32) * 64], 64, 16 * j, lane);
          O[j] = wmma_bf16(pf, bv, O[j]);
        }
      }
    }
    __syncthreads();   // all reads of buffer `cur` done before t+2 overwrites it
  }

  // normalize and write y_attn [B,S,D] as bf16
#pragma unroll
  for (int r = 0; r < 8; ++r) {
    const float rl = 1.0f / li[r];
    const int qg   = qBase + half * 8 + r;
#pragma unroll
    for (int j = 0; j < 4; ++j) {
      Yattn[(size_t)(b * SEQ + qg) * D_MODEL + h * HD + 16 * j + coln] =
          (__bf16)(O[j][r] * rl);
    }
  }
}

// ---------------------------------------------------------------------------
// Kernel 3: out = y_attn @ w_proj^T + b_proj (fp32 output). 64x128 tile.
// ---------------------------------------------------------------------------
__global__ __launch_bounds__(128) void proj_gemm_kernel(
    const __bf16* __restrict__ Y, const float* __restrict__ w_proj,
    const float* __restrict__ b_proj, float* __restrict__ out) {
  __shared__ __align__(16) __bf16 As[64 * 32];
  __shared__ __align__(16) __bf16 Bs[128 * 32];

  const int tid  = threadIdx.x;
  const int lane = tid & 31;
  const int wave = tid >> 5;
  const int gm0  = blockIdx.x * 64;
  const int gn0  = blockIdx.y * 128;

  const int arow = tid >> 1;
  const int acol = (tid & 1) * 16;

  v8f acc[8] = {};

  for (int k0 = 0; k0 < D_MODEL; k0 += 32) {
    const __bf16* aSrc = Y      + (size_t)(gm0 + arow) * D_MODEL + k0 + acol;
    const float*  bSrc = w_proj + (size_t)(gn0 + tid) * D_MODEL + k0;
    if (k0 + 32 < D_MODEL) {
      __builtin_prefetch(aSrc + 32, 0, 1);
      __builtin_prefetch(bSrc + 32, 0, 1);
    }
    // A already bf16: straight 16B copies
    ((uint4*)&As[arow * 32 + acol])[0] = ((const uint4*)aSrc)[0];
    ((uint4*)&As[arow * 32 + acol])[1] = ((const uint4*)aSrc)[1];
    // B fp32 -> bf16 (one full 32-col row per thread)
#pragma unroll
    for (int i = 0; i < 8; ++i) {
      float4 bv = ((const float4*)bSrc)[i];
      const int di = tid * 32 + 4 * i;
      *(unsigned*)&Bs[di]     = pack2(bv.x, bv.y);
      *(unsigned*)&Bs[di + 2] = pack2(bv.z, bv.w);
    }
    __syncthreads();

    const v16bf afrag = load_a_frag(&As[wave * 16 * 32], 32, lane);
#pragma unroll
    for (int j = 0; j < 8; ++j) {
      const v16bf bfrag = load_b_frag(&Bs[j * 16 * 32], 32, lane);
      acc[j] = wmma_bf16(afrag, bfrag, acc[j]);
    }
    __syncthreads();
  }

  const int half = lane >> 4;
  const int coln = lane & 15;
#pragma unroll
  for (int j = 0; j < 8; ++j) {
#pragma unroll
    for (int r = 0; r < 8; ++r) {
      const int gm = gm0 + wave * 16 + half * 8 + r;
      const int gn = gn0 + j * 16 + coln;
      out[(size_t)gm * D_MODEL + gn] = acc[j][r] + b_proj[gn];
    }
  }
}

// ---------------------------------------------------------------------------
extern "C" void kernel_launch(void* const* d_in, const int* in_sizes, int n_in,
                              void* d_out, int out_size, void* d_ws, size_t ws_size,
                              hipStream_t stream) {
  const float* x      = (const float*)d_in[0];
  const float* w_attn = (const float*)d_in[1];
  const float* b_attn = (const float*)d_in[2];
  const float* w_proj = (const float*)d_in[3];
  const float* b_proj = (const float*)d_in[4];
  float* out = (float*)d_out;

  // workspace layout: Q | K | V | Yattn, all bf16 (4 x 8 MB = 32 MB)
  const size_t qkvElems = (size_t)NB * NH * SEQ * HD;
  __bf16* Qws   = (__bf16*)d_ws;
  __bf16* Kws   = Qws + qkvElems;
  __bf16* Vws   = Kws + qkvElems;
  __bf16* Yattn = Vws + qkvElems;

  const dim3 blk(128);
  qkv_gemm_kernel<<<dim3(MROWS / 64, (3 * D_MODEL) / 128), blk, 0, stream>>>(
      x, w_attn, b_attn, Qws, Kws, Vws);
  flash_attn_kernel<<<dim3(SEQ / 64, NH, NB), blk, 0, stream>>>(
      Qws, Kws, Vws, Yattn);
  proj_gemm_kernel<<<dim3(MROWS / 64, D_MODEL / 128), blk, 0, stream>>>(
      Yattn, w_proj, b_proj, out);
}
